// LinearBase_10084583211483
// MI455X (gfx1250) — matrix operand order repaired and checked by reading, not compiled, not run on hardware
//
#include <hip/hip_runtime.h>
#include <hip/hip_bf16.h>

typedef __attribute__((ext_vector_type(16))) __bf16 v16bf;
typedef __attribute__((ext_vector_type(8)))  __bf16 v8bf;
typedef __attribute__((ext_vector_type(8)))  float  v8f;

#define QBLOCK 128   // quantization block size (K and N direction)

#if defined(__has_builtin)
#if __has_builtin(__builtin_amdgcn_global_load_async_to_lds_b128)
#define HAVE_ASYNC_COPY 1
#endif
#if __has_builtin(__builtin_amdgcn_s_wait_asynccnt)
#define HAVE_WAIT_ASYNC 1
#endif
#endif

#if HAVE_ASYNC_COPY
// Matches the builtin's parameter types exactly (per hipcc diagnostic):
//   arg0: __attribute__((vector_size(16))) int  __device__ * (global, AS1)
//   arg1: same pointee in LDS (AS3)
typedef int b128_t __attribute__((vector_size(4 * sizeof(int))));
typedef __attribute__((address_space(1))) b128_t* async_gp_t;
typedef __attribute__((address_space(3))) b128_t* async_lp_t;
#endif

// Convert two float4 chunks (8 floats) into bf16 elements [base..base+7] of frag.
__device__ __forceinline__ void cvt8(v16bf& frag, int base, const float4& lo, const float4& hi) {
    frag[base + 0] = (__bf16)lo.x; frag[base + 1] = (__bf16)lo.y;
    frag[base + 2] = (__bf16)lo.z; frag[base + 3] = (__bf16)lo.w;
    frag[base + 4] = (__bf16)hi.x; frag[base + 5] = (__bf16)hi.y;
    frag[base + 6] = (__bf16)hi.z; frag[base + 7] = (__bf16)hi.w;
}

__device__ __forceinline__ v8bf pack8(const float4& lo, const float4& hi) {
    v8bf p;
    p[0] = (__bf16)lo.x; p[1] = (__bf16)lo.y; p[2] = (__bf16)lo.z; p[3] = (__bf16)lo.w;
    p[4] = (__bf16)hi.x; p[5] = (__bf16)hi.y; p[6] = (__bf16)hi.z; p[7] = (__bf16)hi.w;
    return p;
}

// gemm_a8w8_blockscale:
//   y[m,n] = sum_kb( xs[m,kb] * ws[n/128,kb] * sum_{k in kb} x[m,k]*w[n,k] ) + bias[n]
// WG = 8 wave32s, tile 128(M) x 64(N). Weight tile (64 x 128 f32) is async-staged
// into LDS once per K-block, converted to bf16 once, then all waves feed WMMAs
// from LDS. A fragments are wave-private and come straight from global.
__global__ __launch_bounds__(256)
void gemm_a8w8_blockscale_wmma(const float* __restrict__ x,
                               const float* __restrict__ x_scale,
                               const float* __restrict__ weight,
                               const float* __restrict__ weight_scale,
                               const float* __restrict__ bias,
                               float* __restrict__ out,
                               int M, int N, int K) {
    __shared__ float  lds_w_f32[64 * QBLOCK];    // 32 KB raw staging
    __shared__ __bf16 lds_w_bf16[64 * QBLOCK];   // 16 KB converted tile

    const int KB   = K / QBLOCK;
    const int tid  = threadIdx.x;
    const int lane = tid & 31;
    const int wave = tid >> 5;

    const int m0 = blockIdx.y * 128 + wave * 16;   // wave's M origin
    const int n0 = blockIdx.x * 64;                // WG's N origin (64 weight rows)

    const int lmod = lane & 15;
    const int lhi  = lane >> 4;

    // A fragment addressing (16-bit A 16x32 layout)
    const int rowA  = m0 + lmod;
    const int kAoff = lhi * 8;
    // B fragment: lane<16 -> K 0..7,16..23, lane>=16 -> K 8..15,24..31 of each 32-K step
    const int kBoff = lhi * 8;
    // C/D layout rows
    const int rowC  = m0 + 8 * lhi;

    const int nblk = n0 >> 7;   // 64-wide N span sits inside one 128-wide N block

    // ---- stage issue: move weight[n0 .. n0+64) x [kbase .. kbase+128) into LDS ----
    auto issue_stage = [&](int kbase) {
        const float* gbase = weight + (size_t)n0 * K + kbase;
        #pragma unroll
        for (int c = 0; c < 8; ++c) {               // 8 x b128 per thread
            const int f   = 4 * (c * 256 + tid);    // flat float index in tile
            const int r   = f >> 7;                 // weight row within tile
            const int col = f & (QBLOCK - 1);       // k within block
            const float* g = gbase + (size_t)r * K + col;
#if HAVE_ASYNC_COPY
            __builtin_amdgcn_global_load_async_to_lds_b128(
                (async_gp_t)g,
                (async_lp_t)&lds_w_f32[f],
                0, 0);
#else
            float4 v = *(const float4*)g;
            *(float4*)&lds_w_f32[f] = v;
#endif
        }
    };

    v8f acc[4];
    #pragma unroll
    for (int t = 0; t < 4; ++t) acc[t] = (v8f){0.f,0.f,0.f,0.f,0.f,0.f,0.f,0.f};

    issue_stage(0);

    for (int kb = 0; kb < KB; ++kb) {
        const int kbase = kb * QBLOCK;

        // 1) wait for this block's staged weight data, make it WG-visible
#if HAVE_ASYNC_COPY
#if HAVE_WAIT_ASYNC
        __builtin_amdgcn_s_wait_asynccnt(0);
#else
        asm volatile("s_wait_asynccnt 0x0" ::: "memory");
#endif
#endif
        __syncthreads();

        // 2) cooperative f32 -> bf16 conversion (once per WG, not per wave)
        {
            const int f0 = tid * 32;
            #pragma unroll
            for (int q = 0; q < 4; ++q) {
                const int f = f0 + q * 8;
                float4 lo = *(const float4*)&lds_w_f32[f];
                float4 hi = *(const float4*)&lds_w_f32[f + 4];
                *(v8bf*)&lds_w_bf16[f] = pack8(lo, hi);
            }
        }
        __syncthreads();

        // 3) overlap: kick off async staging of the next K-block
        if (kb + 1 < KB) {
            issue_stage(kbase + QBLOCK);
            __builtin_prefetch(&x[(size_t)rowA * K + kbase + QBLOCK], 0, 3);
        }

        // 4) compute: 4 K-steps x 4 N-tiles of WMMA, B from LDS, A from global
        v8f accq[4];
        #pragma unroll
        for (int t = 0; t < 4; ++t) accq[t] = (v8f){0.f,0.f,0.f,0.f,0.f,0.f,0.f,0.f};

        #pragma unroll
        for (int s = 0; s < 4; ++s) {
            const float* pa = &x[(size_t)rowA * K + kbase + s * 32 + kAoff];
            float4 a_lo0 = *(const float4*)(pa + 0);
            float4 a_hi0 = *(const float4*)(pa + 4);
            float4 a_lo1 = *(const float4*)(pa + 16);
            float4 a_hi1 = *(const float4*)(pa + 20);
            v16bf afrag;
            cvt8(afrag, 0, a_lo0, a_hi0);
            cvt8(afrag, 8, a_lo1, a_hi1);

            #pragma unroll
            for (int t = 0; t < 4; ++t) {
                const __bf16* pb = &lds_w_bf16[(t * 16 + lmod) * QBLOCK + s * 32 + kBoff];
                v8bf blo = *(const v8bf*)pb;        // ds_load_b128
                v8bf bhi = *(const v8bf*)(pb + 16); // ds_load_b128
                v16bf bfrag = __builtin_shufflevector(blo, bhi,
                    0,1,2,3,4,5,6,7,8,9,10,11,12,13,14,15);

                accq[t] = __builtin_amdgcn_wmma_f32_16x16x32_bf16(
                    false, afrag, false, bfrag, (short)0, accq[t],
                    /*reuse_a=*/false, /*reuse_b=*/false);
            }
        }

        // 5) fold block scales: s(m,n) = x_scale[m,kb] * weight_scale[n/128,kb]
        const float ws = weight_scale[(size_t)nblk * KB + kb];
        float svec[8];
        #pragma unroll
        for (int v = 0; v < 8; ++v)
            svec[v] = ws * x_scale[(size_t)(rowC + v) * KB + kb];

        #pragma unroll
        for (int t = 0; t < 4; ++t) {
            #pragma unroll
            for (int v = 0; v < 8; ++v)
                acc[t][v] = fmaf(svec[v], accq[t][v], acc[t][v]);
        }
    }

    // ---- epilogue: fused bias, write 16x64 strip per wave ----
    #pragma unroll
    for (int t = 0; t < 4; ++t) {
        const int coln = n0 + t * 16 + lmod;
        const float b = bias[coln];
        #pragma unroll
        for (int v = 0; v < 8; ++v) {
            out[(size_t)(rowC + v) * N + coln] = acc[t][v] + b;
        }
    }
}

extern "C" void kernel_launch(void* const* d_in, const int* in_sizes, int n_in,
                              void* d_out, int out_size, void* d_ws, size_t ws_size,
                              hipStream_t stream) {
    const float* x            = (const float*)d_in[0];
    const float* x_scale      = (const float*)d_in[1];
    const float* weight       = (const float*)d_in[2];
    const float* weight_scale = (const float*)d_in[3];
    const float* bias         = (const float*)d_in[4];
    float*       out          = (float*)d_out;

    // Dims: in_sizes[0] = M*K, in_sizes[1] = M*(K/128), in_sizes[4] = N
    const long long xk = in_sizes[0];
    const long long xs = in_sizes[1];
    const int K = (int)(128LL * (xk / xs));
    const int M = (int)(xk / K);
    const int N = in_sizes[4];

    dim3 block(256, 1, 1);
    dim3 grid((unsigned)(N / 64), (unsigned)(M / 128), 1);
    gemm_a8w8_blockscale_wmma<<<grid, block, 0, stream>>>(
        x, x_scale, weight, weight_scale, bias, out, M, N, K);
}
